// CQAttention_19473381720742
// MI455X (gfx1250) — matrix operand order repaired and checked by reading, not compiled, not run on hardware
//
#include <hip/hip_runtime.h>
#include <hip/hip_bf16.h>

#define BB 32
#define DD 256
#define LC 2048
#define LQ 256
#define NEG_INF (-1e30f)

typedef __attribute__((ext_vector_type(16))) __bf16 v16bf;
typedef __attribute__((ext_vector_type(8)))  float  v8f;

__device__ __forceinline__ unsigned short f2bf(float f) {
  unsigned int u = __float_as_uint(f);
  u += 0x7FFFu + ((u >> 16) & 1u);          // round-to-nearest-even
  return (unsigned short)(u >> 16);
}

// A-fragment (16xK=32, bf16) gathered from row-major LDS tile [row][k].
// Per ISA 7.12.2: lane m = lane&15; VGPR v<4 holds K pairs {2v, 2v+1} + 8*hi,
// VGPR v>=4 holds {16+2(v&3), +1} + 8*hi  (hi = lane>>4).
__device__ __forceinline__ v16bf frag_a(const unsigned short* base, int stride,
                                        int row0, int kbase, int lane) {
  const int l15 = lane & 15, hi = lane >> 4;
  union { unsigned int u[8]; v16bf v; } fr;
  const unsigned short* rp = base + (row0 + l15) * stride;
#pragma unroll
  for (int t = 0; t < 8; ++t) {
    int k = kbase + ((t < 4) ? 0 : 16) + hi * 8 + 2 * (t & 3);
    fr.u[t] = *(const unsigned int*)(rp + k);
  }
  return fr.v;
}

// B-fragment (K=32 x 16, bf16) gathered from an LDS tile stored TRANSPOSED
// as [n][k]: lane n = lane&15; VGPR v<4 holds K pair {2v, 2v+1} + 8*hi,
// VGPR v>=4 holds {16+2(v&3), +1} + 8*hi  (hi = lane>>4).
__device__ __forceinline__ v16bf frag_b(const unsigned short* base, int stride,
                                        int n0, int kbase, int lane) {
  const int l15 = lane & 15, hi = lane >> 4;
  union { unsigned int u[8]; v16bf v; } fr;
  const unsigned short* rp = base + (n0 + l15) * stride;
#pragma unroll
  for (int t = 0; t < 8; ++t) {
    int k = kbase + ((t < 4) ? 0 : 16) + hi * 8 + 2 * (t & 3);
    fr.u[t] = *(const unsigned int*)(rp + k);
  }
  return fr.v;
}

// ---------------- cw[b,c] = Ct@w1 ; qw[b,q] = Qt@w2 ----------------
__global__ __launch_bounds__(256) void k_vecw(const float* __restrict__ C,
                                              const float* __restrict__ Q,
                                              const float* __restrict__ w,
                                              float* __restrict__ cw,
                                              float* __restrict__ qw) {
  int idx = blockIdx.x * 256 + threadIdx.x;
  if (idx < BB * LC) {
    int b = idx / LC, c = idx % LC;
    const float* Cp = C + (size_t)b * DD * LC + c;
    float s = 0.f;
#pragma unroll 4
    for (int d = 0; d < DD; ++d) s += Cp[(size_t)d * LC] * w[d];
    cw[idx] = s;
  } else if (idx < BB * LC + BB * LQ) {
    int j = idx - BB * LC;
    int b = j / LQ, q = j % LQ;
    const float* Qp = Q + (size_t)b * DD * LQ + q;
    float s = 0.f;
#pragma unroll 4
    for (int d = 0; d < DD; ++d) s += Qp[(size_t)d * LQ] * w[DD + d];
    qw[j] = s;
  }
}

// ---------------- S = cw + qw + (Ct*w3) @ Qt^T  (bf16 WMMA) ----------------
// Block tile: 32 c x 64 q, K = 256 fully staged. 8 waves = 2x4 subtiles.
__global__ __launch_bounds__(256) void k_sgemm(const float* __restrict__ C,
                                               const float* __restrict__ Q,
                                               const float* __restrict__ w,
                                               const float* __restrict__ cw,
                                               const float* __restrict__ qw,
                                               float* __restrict__ S) {
  __shared__ unsigned short lA[32][264];   // [c][d]  A = (Ct*w3)
  __shared__ unsigned short lB[64][264];   // [q][d]  B^T (B = Qt^T: [d][q])
  const int tid = threadIdx.x;
  const int b  = blockIdx.z;
  const int c0 = blockIdx.y * 32;
  const int q0 = blockIdx.x * 64;
  const size_t cbase = (size_t)b * DD * LC;
  const size_t qbase = (size_t)b * DD * LQ;

  for (int i = tid; i < 32 * 256; i += 256) {
    int cc = i & 31, d = i >> 5;
    lA[cc][d] = f2bf(C[cbase + (size_t)d * LC + c0 + cc] * w[2 * DD + d]);
  }
  for (int i = tid; i < 64 * 256; i += 256) {
    int q = i & 63, d = i >> 6;
    lB[q][d] = f2bf(Q[qbase + (size_t)d * LQ + q0 + q]);
  }
  __syncthreads();

  const int wave = tid >> 5, lane = tid & 31;
  const int m0 = (wave >> 2) * 16, n0 = (wave & 3) * 16;
  v8f acc = {0.f, 0.f, 0.f, 0.f, 0.f, 0.f, 0.f, 0.f};
#pragma unroll
  for (int kb = 0; kb < 8; ++kb) {
    v16bf a  = frag_a(&lA[0][0], 264, m0, kb * 32, lane);
    v16bf bm = frag_b(&lB[0][0], 264, n0, kb * 32, lane);
    acc = __builtin_amdgcn_wmma_f32_16x16x32_bf16(false, a, false, bm,
                                                  (short)0, acc, false, false);
  }
  const int hi = lane >> 4, l15 = lane & 15;
  const int n = q0 + n0 + l15;
  const float qwv = qw[b * LQ + n];
  float* Sp = S + (size_t)b * LC * LQ;
#pragma unroll
  for (int r = 0; r < 8; ++r) {
    int m = c0 + m0 + r + hi * 8;
    Sp[(size_t)m * LQ + n] = acc[r] + cw[b * LC + m] + qwv;
  }
}

// ---------------- row softmax stats (over q) for S1 ----------------
__global__ __launch_bounds__(256) void k_rowstats(const float* __restrict__ S,
                                                  const float* __restrict__ qmask,
                                                  float* __restrict__ rmax,
                                                  float* __restrict__ rsum) {
  const int wave = threadIdx.x >> 5, lane = threadIdx.x & 31;
  const int row = blockIdx.x * 8 + wave;          // b*LC + c
  const int b = row / LC;
  const float* Sp = S + (size_t)row * LQ;
  const float* qm = qmask + b * LQ;
  float m = -3.0e38f, s = 0.f;
#pragma unroll
  for (int j = 0; j < 8; ++j) {
    int q = lane + j * 32;
    float qv = qm[q];
    float v = Sp[q] * qv + (1.f - qv) * NEG_INF;
    float nm = fmaxf(m, v);
    s = s * __expf(m - nm) + __expf(v - nm);
    m = nm;
  }
#pragma unroll
  for (int off = 16; off > 0; off >>= 1) {
    float om = __shfl_xor(m, off);
    float os = __shfl_xor(s, off);
    float nm = fmaxf(m, om);
    s = s * __expf(m - nm) + os * __expf(om - nm);
    m = nm;
  }
  if (lane == 0) { rmax[row] = m; rsum[row] = s; }
}

// ---------------- column softmax stats (over c) for S2, two-phase ----------------
__global__ __launch_bounds__(256) void k_colstats_part(const float* __restrict__ S,
                                                       const float* __restrict__ cmask,
                                                       float* __restrict__ pmax,
                                                       float* __restrict__ psum) {
  const int chunk = blockIdx.x, b = blockIdx.y, q = threadIdx.x;
  const float* Sp = S + (size_t)b * LC * LQ + q;
  float m = -3.0e38f, s = 0.f;
  for (int j = 0; j < 256; ++j) {
    int c = chunk * 256 + j;
    float cm = cmask[b * LC + c];
    float v = Sp[(size_t)c * LQ] * cm + (1.f - cm) * NEG_INF;
    float nm = fmaxf(m, v);
    s = s * __expf(m - nm) + __expf(v - nm);
    m = nm;
  }
  int o = (chunk * BB + b) * LQ + q;
  pmax[o] = m; psum[o] = s;
}

__global__ __launch_bounds__(256) void k_colstats_comb(const float* __restrict__ pmax,
                                                       const float* __restrict__ psum,
                                                       float* __restrict__ cmax,
                                                       float* __restrict__ csum) {
  int idx = blockIdx.x * 256 + threadIdx.x;       // b*LQ + q
  if (idx >= BB * LQ) return;
  int b = idx / LQ, q = idx % LQ;
  float m = -3.0e38f, s = 0.f;
#pragma unroll
  for (int chunk = 0; chunk < 8; ++chunk) {
    int o = (chunk * BB + b) * LQ + q;
    float om = pmax[o], os = psum[o];
    float nm = fmaxf(m, om);
    s = s * __expf(m - nm) + os * __expf(om - nm);
    m = nm;
  }
  cmax[idx] = m; csum[idx] = s;
}

// ---------------- T[b,d,q] = sum_k C[b,d,k] * S2[k,q]  -> bf16 ----------------
// Block tile 32 d x 64 q, K = 2048 in chunks of 128. S2 regenerated on the fly.
__global__ __launch_bounds__(256) void k_tgemm(const float* __restrict__ C,
                                               const float* __restrict__ S,
                                               const float* __restrict__ cmask,
                                               const float* __restrict__ cmax,
                                               const float* __restrict__ csum,
                                               unsigned short* __restrict__ Tb) {
  __shared__ unsigned short lA[32][136];   // [d][k]
  __shared__ unsigned short lB[64][136];   // [q][k] (B transposed)
  const int tid = threadIdx.x;
  const int b  = blockIdx.z;
  const int d0 = blockIdx.y * 32;
  const int q0 = blockIdx.x * 64;
  const size_t cbase = (size_t)b * DD * LC;
  const float* Sp = S + (size_t)b * LC * LQ;
  const int wave = tid >> 5, lane = tid & 31;
  const int m0 = (wave >> 2) * 16, n0 = (wave & 3) * 16;
  v8f acc = {0.f, 0.f, 0.f, 0.f, 0.f, 0.f, 0.f, 0.f};

  for (int kc = 0; kc < 16; ++kc) {
    const int k0 = kc * 128;
    __syncthreads();
    {
      // prefetch next K-chunk of C while staging current one
      int k = tid & 127, dd = tid >> 7;
      if (kc < 15)
        __builtin_prefetch(&C[cbase + (size_t)(d0 + dd) * LC + k0 + 128 + k], 0, 1);
    }
    for (int i = tid; i < 32 * 128; i += 256) {
      int k = i & 127, dd = i >> 7;
      lA[dd][k] = f2bf(C[cbase + (size_t)(d0 + dd) * LC + k0 + k]);
    }
    for (int i = tid; i < 64 * 128; i += 256) {
      int q = i & 63, k = i >> 6;
      int krow = k0 + k;
      float cm = cmask[b * LC + krow];
      float v = Sp[(size_t)krow * LQ + q0 + q] * cm + (1.f - cm) * NEG_INF;
      int qq = b * LQ + q0 + q;
      lB[q][k] = f2bf(__expf(v - cmax[qq]) / csum[qq]);
    }
    __syncthreads();
#pragma unroll
    for (int kb = 0; kb < 4; ++kb) {
      v16bf a  = frag_a(&lA[0][0], 136, m0, kb * 32, lane);
      v16bf bm = frag_b(&lB[0][0], 136, n0, kb * 32, lane);
      acc = __builtin_amdgcn_wmma_f32_16x16x32_bf16(false, a, false, bm,
                                                    (short)0, acc, false, false);
    }
  }
  const int hi = lane >> 4, l15 = lane & 15;
  const int n = q0 + n0 + l15;
#pragma unroll
  for (int r = 0; r < 8; ++r) {
    int m = d0 + m0 + r + hi * 8;
    Tb[(size_t)b * DD * LQ + (size_t)m * LQ + n] = f2bf(acc[r]);
  }
}

// ------- out = [Ct, A, Ct*A, Ct*Bv]^T via [Q;T]^T @ S1^T, fused epilogue -------
// Block tile 64 d' x 32 c (d' in [0,512): 0..255 -> A-part, 256..511 -> Bv-part).
__global__ __launch_bounds__(256) void k_outgemm(const float* __restrict__ C,
                                                 const float* __restrict__ Q,
                                                 const float* __restrict__ S,
                                                 const float* __restrict__ qmask,
                                                 const float* __restrict__ rmax,
                                                 const float* __restrict__ rsum,
                                                 const unsigned short* __restrict__ Tb,
                                                 float* __restrict__ out) {
  __shared__ unsigned short lA[64][264];   // [d'][q]
  __shared__ unsigned short lB[32][264];   // [c][q]  (S1^T transposed)
  const int tid = threadIdx.x;
  const int b   = blockIdx.z;
  const int dp0 = blockIdx.y * 64;
  const int c0  = blockIdx.x * 32;
  const float* Sp = S + (size_t)b * LC * LQ;

  if (dp0 < DD) {
    const float* Qp = Q + (size_t)b * DD * LQ;
    for (int i = tid; i < 64 * 256; i += 256) {
      int q = i & 255, dd = i >> 8;
      lA[dd][q] = f2bf(Qp[(size_t)(dp0 + dd) * LQ + q]);
    }
  } else {
    // Tb tile is already bf16: stage it with the CDNA5 async global->LDS DMA
    // path (no VGPR round trip, tracked by ASYNCcnt). 16 B per lane per issue;
    // each 32-lane group moves one full 512 B row.
    const unsigned short* Tp = Tb + (size_t)b * DD * LQ + (size_t)(dp0 - DD) * LQ;
    const unsigned lds_base = (unsigned)(unsigned long long)&lA[0][0];
#pragma unroll
    for (int j = 0; j < 8; ++j) {
      int flat = (j * 256 + tid) * 8;             // ushort index within tile
      int q = flat & 255;
      int dd = flat >> 8;
      unsigned lds_addr = lds_base + (unsigned)(dd * 264 + q) * 2u;
      unsigned long long ga = (unsigned long long)(Tp + (size_t)dd * LQ + q);
      asm volatile("global_load_async_to_lds_b128 %0, %1, off"
                   :: "v"(lds_addr), "v"(ga) : "memory");
    }
    asm volatile("s_wait_asynccnt 0x0" ::: "memory");
  }
  for (int i = tid; i < 32 * 256; i += 256) {
    int q = i & 255, cc = i >> 8;
    int c = c0 + cc;
    float qv = qmask[b * LQ + q];
    float v = Sp[(size_t)c * LQ + q] * qv + (1.f - qv) * NEG_INF;
    lB[cc][q] = f2bf(__expf(v - rmax[b * LC + c]) / rsum[b * LC + c]);
  }
  __syncthreads();

  const int wave = tid >> 5, lane = tid & 31;
  const int m0 = (wave >> 1) * 16, n0 = (wave & 1) * 16;
  v8f acc = {0.f, 0.f, 0.f, 0.f, 0.f, 0.f, 0.f, 0.f};
#pragma unroll
  for (int kb = 0; kb < 8; ++kb) {
    v16bf a  = frag_a(&lA[0][0], 264, m0, kb * 32, lane);
    v16bf bm = frag_b(&lB[0][0], 264, n0, kb * 32, lane);
    acc = __builtin_amdgcn_wmma_f32_16x16x32_bf16(false, a, false, bm,
                                                  (short)0, acc, false, false);
  }
  const int hi = lane >> 4, l15 = lane & 15;
  const int c = c0 + n0 + l15;
  float* ob = out + (size_t)b * 4 * DD * LC;
  const float* Cb = C + (size_t)b * DD * LC;
#pragma unroll
  for (int r = 0; r < 8; ++r) {
    int dp = dp0 + m0 + r + hi * 8;
    float av = acc[r];
    if (dp < DD) {
      float cv = Cb[(size_t)dp * LC + c];
      ob[(size_t)dp * LC + c]            = cv;         // Ct
      ob[(size_t)(DD + dp) * LC + c]     = av;         // A
      ob[(size_t)(2 * DD + dp) * LC + c] = cv * av;    // Ct*A
    } else {
      int d = dp - DD;
      float cv = Cb[(size_t)d * LC + c];
      ob[(size_t)(3 * DD + d) * LC + c]  = cv * av;    // Ct*Bv
    }
  }
}

extern "C" void kernel_launch(void* const* d_in, const int* in_sizes, int n_in,
                              void* d_out, int out_size, void* d_ws, size_t ws_size,
                              hipStream_t stream) {
  const float* C     = (const float*)d_in[0];
  const float* Q     = (const float*)d_in[1];
  const float* cmask = (const float*)d_in[2];
  const float* qmask = (const float*)d_in[3];
  const float* w     = (const float*)d_in[4];
  float* out = (float*)d_out;

  char* p = (char*)d_ws;
  float* S    = (float*)p;  p += (size_t)BB * LC * LQ * 4;   // 64 MB
  float* rmax = (float*)p;  p += (size_t)BB * LC * 4;
  float* rsum = (float*)p;  p += (size_t)BB * LC * 4;
  float* cmax = (float*)p;  p += (size_t)BB * LQ * 4;
  float* csum = (float*)p;  p += (size_t)BB * LQ * 4;
  float* pmax = (float*)p;  p += (size_t)8 * BB * LQ * 4;
  float* psum = (float*)p;  p += (size_t)8 * BB * LQ * 4;
  float* cw   = (float*)p;  p += (size_t)BB * LC * 4;
  float* qw   = (float*)p;  p += (size_t)BB * LQ * 4;
  unsigned short* Tb = (unsigned short*)p;  p += (size_t)BB * DD * LQ * 2;

  k_vecw<<<dim3((BB * LC + BB * LQ + 255) / 256), 256, 0, stream>>>(C, Q, w, cw, qw);
  k_sgemm<<<dim3(LQ / 64, LC / 32, BB), 256, 0, stream>>>(C, Q, w, cw, qw, S);
  k_rowstats<<<dim3(BB * LC / 8), 256, 0, stream>>>(S, qmask, rmax, rsum);
  k_colstats_part<<<dim3(8, BB), 256, 0, stream>>>(S, cmask, pmax, psum);
  k_colstats_comb<<<dim3((BB * LQ + 255) / 256), 256, 0, stream>>>(pmax, psum, cmax, csum);
  k_tgemm<<<dim3(LQ / 64, DD / 32, BB), 256, 0, stream>>>(C, S, cmask, cmax, csum, Tb);
  k_outgemm<<<dim3(LC / 32, 2 * DD / 64, BB), 256, 0, stream>>>(C, Q, S, qmask, rmax, rsum, Tb, out);
}
